// GraphAttentionLayer_32023276159596
// MI455X (gfx1250) — matrix operand order repaired and checked by reading, not compiled, not run on hardware
//
#include <hip/hip_runtime.h>

typedef __attribute__((ext_vector_type(16))) _Float16 v16h;
typedef __attribute__((ext_vector_type(2)))  _Float16 v2h;
typedef __attribute__((ext_vector_type(8)))  float    v8f;
typedef __attribute__((ext_vector_type(2)))  float    v2f;
typedef __attribute__((ext_vector_type(4)))  int      v4i;

#define BATCH   8
#define NNODE   2048
#define FDIM    128
#define NEG_INF (-1000000000000.0f)
#define ROWH    40   // LDS row stride in halfs (80 B: 16B-aligned, bank-conflict free)

__device__ __forceinline__ float lrelu(float x) { return x > 0.f ? x : 0.01f * x; }

// ---- CDNA5 async global->LDS copy (ASYNCcnt) with asm fallback ------------
__device__ __forceinline__ void async_g2l_b128(const _Float16* g, _Float16* l) {
#if __has_builtin(__builtin_amdgcn_global_load_async_to_lds_b128)
    __builtin_amdgcn_global_load_async_to_lds_b128(
        (__attribute__((address_space(1))) v4i*)const_cast<_Float16*>(g),
        (__attribute__((address_space(3))) v4i*)l,
        0, 0);
#else
    __attribute__((address_space(3))) void* lp = (__attribute__((address_space(3))) void*)l;
    asm volatile("global_load_async_to_lds_b128 %0, %1, off"
                 :: "v"(lp), "v"(g) : "memory");
#endif
}
__device__ __forceinline__ void wait_async_le2() {
#if __has_builtin(__builtin_amdgcn_s_wait_asynccnt)
    __builtin_amdgcn_s_wait_asynccnt(2);
#else
    asm volatile("s_wait_asynccnt 0x2" ::: "memory");
#endif
}
__device__ __forceinline__ void wait_async_le0() {
#if __has_builtin(__builtin_amdgcn_s_wait_asynccnt)
    __builtin_amdgcn_s_wait_asynccnt(0);
#else
    asm volatile("s_wait_asynccnt 0x0" ::: "memory");
#endif
}

// ---------------------------------------------------------------------------
// Kernel 1: h = inp @ W via V_WMMA_F32_16X16X4_F32 (exact fp32).
// Also computes s1 = h@a1, s2 = h@a2, and stores h transposed to f16:
//   hT[b][f][j]  (feature-major) so kernel2 B-fragments are contiguous.
// ---------------------------------------------------------------------------
__global__ __launch_bounds__(256) void gat_h_kernel(
    const float* __restrict__ inp, const float* __restrict__ W,
    const float* __restrict__ a,   _Float16* __restrict__ hT,
    float* __restrict__ s1, float* __restrict__ s2)
{
    const int wave = blockIdx.x * (blockDim.x >> 5) + (threadIdx.x >> 5);
    const int lane = threadIdx.x & 31;
    const int b    = wave >> 7;            // 128 row tiles per batch
    const int i0   = (wave & 127) << 4;
    const int lm   = lane & 15;
    const int g    = lane >> 4;

    const float* Arow = inp + ((size_t)b * NNODE + (i0 + lm)) * FDIM;

    v8f acc[8] = {};
    for (int kk = 0; kk < FDIM; kk += 4) {
        const int k0 = kk + 2 * g;
        v2f av;
        av.x = Arow[k0];
        av.y = Arow[k0 + 1];
#pragma unroll
        for (int nt = 0; nt < 8; ++nt) {
            const int n = nt * 16 + lm;
            v2f bv;
            bv.x = W[(size_t)k0 * FDIM + n];
            bv.y = W[(size_t)(k0 + 1) * FDIM + n];
            acc[nt] = __builtin_amdgcn_wmma_f32_16x16x4_f32(
                false, av, false, bv, (short)0, acc[nt], false, false);
        }
    }

    float p1[8] = {}, p2[8] = {};
#pragma unroll
    for (int nt = 0; nt < 8; ++nt) {
        const int f = nt * 16 + lm;
        const float a1v = a[f];
        const float a2v = a[FDIM + f];
#pragma unroll
        for (int r = 0; r < 8; ++r) {
            p1[r] += acc[nt][r] * a1v;
            p2[r] += acc[nt][r] * a2v;
        }
    }
#pragma unroll
    for (int r = 0; r < 8; ++r) {
#pragma unroll
        for (int m = 1; m < 16; m <<= 1) {
            p1[r] += __shfl_xor(p1[r], m, 32);
            p2[r] += __shfl_xor(p2[r], m, 32);
        }
    }
    if (lm == 0) {
        const size_t base = (size_t)b * NNODE + i0 + 8 * g;
#pragma unroll
        for (int r = 0; r < 8; ++r) {
            s1[base + r] = p1[r];
            s2[base + r] = p2[r];
        }
    }

#pragma unroll
    for (int nt = 0; nt < 8; ++nt) {
        const int f = nt * 16 + lm;
        _Float16* dst = hT + ((size_t)b * FDIM + f) * NNODE + i0 + 8 * g;
#pragma unroll
        for (int r = 0; r < 8; r += 2) {
            v2h pk;
            pk.x = (_Float16)acc[nt][r];
            pk.y = (_Float16)acc[nt][r + 1];
            *(v2h*)(dst + r) = pk;
        }
    }
}

// ---------------------------------------------------------------------------
// Kernel 2: flash-attention-style masked softmax + (attn @ h) via
// V_WMMA_F32_16X16X32_F16. One wave per 16-row tile; 8 waves per block share
// the same batch, so the h tile (128 feats x 32 nodes, f16 = 8 KB) is staged
// once per block through double-buffered async global->LDS copies.
// ---------------------------------------------------------------------------
__global__ __launch_bounds__(256) void gat_attn_kernel(
    const int* __restrict__ adj, const float* __restrict__ s1g,
    const float* __restrict__ s2g, const _Float16* __restrict__ hT,
    float* __restrict__ out)
{
    __shared__ _Float16 lbuf[2][FDIM * ROWH];

    const int tid  = threadIdx.x;
    const int wave = blockIdx.x * (blockDim.x >> 5) + (tid >> 5);
    const int lane = tid & 31;
    const int b    = wave >> 7;            // uniform across the block
    const int i0   = (wave & 127) << 4;
    const int lm   = lane & 15;
    const int g    = lane >> 4;

    // Cooperative loader mapping: thread -> 32 B of one hT row.
    const int lf = tid >> 1;               // feature row 0..127
    const int lp = (tid & 1) * 16;         // half-offset within 32-node row
    const _Float16* gsrc = hT + ((size_t)b * FDIM + lf) * NNODE + lp;
    _Float16* ldst0 = &lbuf[0][lf * ROWH + lp];
    _Float16* ldst1 = &lbuf[1][lf * ROWH + lp];

    const float  s1v  = s1g[(size_t)b * NNODE + i0 + lm];
    const float* s2b  = s2g + (size_t)b * NNODE;
    const int*   adjb = adj + ((size_t)b * NNODE + (i0 + lm)) * (size_t)NNODE;

    float m = -INFINITY;
    float l = 0.0f;
    v8f acc[8] = {};

    // Prologue: stage tile 0 into buffer 0.
    async_g2l_b128(gsrc, ldst0);
    async_g2l_b128(gsrc + 8, ldst0 + 8);

#define EDGE(av_, sv_) (((av_) > 0) ? lrelu(s1v + (sv_)) : NEG_INF)

    for (int j0 = 0; j0 < NNODE; j0 += 32) {
        const int  buf      = (j0 >> 5) & 1;
        const bool has_next = (j0 + 32) < NNODE;
        if (has_next) {                    // stage tile t+1 into the other buffer
            const _Float16* gn = gsrc + j0 + 32;
            _Float16* ln = buf ? ldst0 : ldst1;
            async_g2l_b128(gn, ln);
            async_g2l_b128(gn + 8, ln + 8);
            wait_async_le2();              // our tile-t copies are done
        } else {
            wait_async_le0();
        }
        __syncthreads();                   // everyone's tile-t copies are done

        if (has_next)
            __builtin_prefetch(adjb + j0 + 32 + 8 * g, 0, 1);  // global_prefetch_b8

        const int base1 = j0 + 8 * g;
        const int base2 = base1 + 16;
        const int4   adA = *(const int4*)(adjb + base1);
        const int4   adB = *(const int4*)(adjb + base1 + 4);
        const int4   adC = *(const int4*)(adjb + base2);
        const int4   adD = *(const int4*)(adjb + base2 + 4);
        const float4 sA  = *(const float4*)(s2b + base1);
        const float4 sB  = *(const float4*)(s2b + base1 + 4);
        const float4 sC  = *(const float4*)(s2b + base2);
        const float4 sD  = *(const float4*)(s2b + base2 + 4);

        const float e0  = EDGE(adA.x, sA.x), e1  = EDGE(adA.y, sA.y);
        const float e2  = EDGE(adA.z, sA.z), e3  = EDGE(adA.w, sA.w);
        const float e4  = EDGE(adB.x, sB.x), e5  = EDGE(adB.y, sB.y);
        const float e6  = EDGE(adB.z, sB.z), e7  = EDGE(adB.w, sB.w);
        const float e8  = EDGE(adC.x, sC.x), e9  = EDGE(adC.y, sC.y);
        const float e10 = EDGE(adC.z, sC.z), e11 = EDGE(adC.w, sC.w);
        const float e12 = EDGE(adD.x, sD.x), e13 = EDGE(adD.y, sD.y);
        const float e14 = EDGE(adD.z, sD.z), e15 = EDGE(adD.w, sD.w);

        float tm = fmaxf(fmaxf(fmaxf(e0, e1), fmaxf(e2, e3)),
                         fmaxf(fmaxf(e4, e5), fmaxf(e6, e7)));
        tm = fmaxf(tm, fmaxf(fmaxf(fmaxf(e8, e9), fmaxf(e10, e11)),
                             fmaxf(fmaxf(e12, e13), fmaxf(e14, e15))));
        tm = fmaxf(tm, __shfl_xor(tm, 16, 32));
        const float mn    = fmaxf(m, tm);
        const float scale = __expf(m - mn);

        v16h P;
        float rs = 0.0f;
        {
            float p;
            p = __expf(e0  - mn); rs += p; P[0]  = (_Float16)p;
            p = __expf(e1  - mn); rs += p; P[1]  = (_Float16)p;
            p = __expf(e2  - mn); rs += p; P[2]  = (_Float16)p;
            p = __expf(e3  - mn); rs += p; P[3]  = (_Float16)p;
            p = __expf(e4  - mn); rs += p; P[4]  = (_Float16)p;
            p = __expf(e5  - mn); rs += p; P[5]  = (_Float16)p;
            p = __expf(e6  - mn); rs += p; P[6]  = (_Float16)p;
            p = __expf(e7  - mn); rs += p; P[7]  = (_Float16)p;
            p = __expf(e8  - mn); rs += p; P[8]  = (_Float16)p;
            p = __expf(e9  - mn); rs += p; P[9]  = (_Float16)p;
            p = __expf(e10 - mn); rs += p; P[10] = (_Float16)p;
            p = __expf(e11 - mn); rs += p; P[11] = (_Float16)p;
            p = __expf(e12 - mn); rs += p; P[12] = (_Float16)p;
            p = __expf(e13 - mn); rs += p; P[13] = (_Float16)p;
            p = __expf(e14 - mn); rs += p; P[14] = (_Float16)p;
            p = __expf(e15 - mn); rs += p; P[15] = (_Float16)p;
        }
        rs += __shfl_xor(rs, 16, 32);
        l = l * scale + rs;

        // Rescale accumulators only when some row max actually moved
        // (steady state: skip -> inner loop is loads + exps + WMMA).
        if (__ballot(mn > m)) {
            float srow[8];
#pragma unroll
            for (int r = 0; r < 8; ++r)
                srow[r] = __shfl(scale, r + (g << 3), 32);
#pragma unroll
            for (int nt = 0; nt < 8; ++nt)
#pragma unroll
                for (int r = 0; r < 8; ++r)
                    acc[nt][r] *= srow[r];
        }
        m = mn;

#pragma unroll
        for (int nt = 0; nt < 8; ++nt) {
            typedef __attribute__((ext_vector_type(8))) _Float16 v8h_;
            const _Float16* brow = &lbuf[buf][(nt * 16 + lm) * ROWH + 8 * g];
            const v8h_ blo = *(const v8h_*)brow;
            const v8h_ bhi = *(const v8h_*)(brow + 16);
            const v16h Bf = __builtin_shufflevector(blo, bhi, 0,1,2,3,4,5,6,7,8,9,10,11,12,13,14,15);
            acc[nt] = __builtin_amdgcn_wmma_f32_16x16x32_f16(
                false, P, false, Bf, (short)0, acc[nt], false, false);
        }

        __syncthreads();                   // release buf before it is re-staged
    }
#undef EDGE

    float inv[8];
#pragma unroll
    for (int r = 0; r < 8; ++r)
        inv[r] = 1.0f / __shfl(l, r + (g << 3), 32);

#pragma unroll
    for (int nt = 0; nt < 8; ++nt) {
        const int f = nt * 16 + lm;
#pragma unroll
        for (int r = 0; r < 8; ++r) {
            const int row = i0 + r + 8 * g;
            out[((size_t)b * NNODE + row) * FDIM + f] = acc[nt][r] * inv[r];
        }
    }
}

extern "C" void kernel_launch(void* const* d_in, const int* in_sizes, int n_in,
                              void* d_out, int out_size, void* d_ws, size_t ws_size,
                              hipStream_t stream) {
    const float* inp = (const float*)d_in[0];   // (8,2048,128)
    const int*   adj = (const int*)d_in[1];     // (8,2048,2048)
    const float* W   = (const float*)d_in[2];   // (128,128)
    const float* a   = (const float*)d_in[3];   // (256,1)
    float* out = (float*)d_out;                 // (8,2048,128)

    _Float16* hT = (_Float16*)d_ws;             // 4 MB
    float* s1 = (float*)((char*)d_ws + (size_t)BATCH * FDIM * NNODE * sizeof(_Float16));
    float* s2 = s1 + (size_t)BATCH * NNODE;

    const int waves = BATCH * (NNODE / 16);     // 1024 row tiles
    const int wpb   = 8;
    dim3 block(32 * wpb);
    dim3 grid(waves / wpb);                     // 128 blocks

    gat_h_kernel<<<grid, block, 0, stream>>>(inp, W, a, hT, s1, s2);
    gat_attn_kernel<<<grid, block, 0, stream>>>(adj, s1, s2, hT, out);
}
